// LSTM_89455578841344
// MI455X (gfx1250) — matrix operand order repaired and checked by reading, not compiled, not run on hardware
//
#include <hip/hip_runtime.h>

// LSTM (T=200000, H=100, in=1, out=1) — latency-bound sequential recurrence.
// One workgroup of 13 wave32s (416 threads) on one WGP.
//  - W_hh held in VGPRs in v_wmma_f32_16x16x32_f16 A-layout (f16): each wave owns
//    TWO 16-row gate tiles (rows 32w..32w+31), K=100 padded to 128.
//  - Each step: 8 ds_load_b128 build all four B chunks (h broadcast-replicated
//    across the 16 B columns); a sched_barrier pins them ahead of the WMMAs so a
//    single dscnt wait covers all LDS latency; then 8 WMMAs as two independent
//    accumulation chains (pipelines the matrix unit).
//  - Threads 0..99 run the pointwise cell update with c_j register-resident for
//    all 200000 steps; x[t+1] is prefetched (compiler lowers to scalar load).
//  - Two split barriers per step — the minimum for this dataflow.

typedef __attribute__((ext_vector_type(16))) _Float16 v16h;
typedef __attribute__((ext_vector_type(8)))  float    v8f;
typedef __attribute__((ext_vector_type(4)))  float    v4f;

#define HID       100
#define NGATES    400
#define NGATESPAD 416          // 13 waves * 32 rows (rows 400..415 are zero padding)
#define KPAD      128
#define NWAVES    13
#define NTHREADS  (NWAVES * 32)

__device__ __forceinline__ float fsigmoid(float x) {
    return __builtin_amdgcn_rcpf(1.0f + __expf(-x));       // v_exp_f32 + v_rcp_f32
}
__device__ __forceinline__ float ftanh(float x) {
    return 1.0f - 2.0f * __builtin_amdgcn_rcpf(1.0f + __expf(2.0f * x));
}

__global__ void __launch_bounds__(NTHREADS, 1)
lstm_seq_kernel(const float* __restrict__ x_seq, int T,
                const float* __restrict__ W_ih, const float* __restrict__ W_hh,
                const float* __restrict__ b_ih, const float* __restrict__ b_hh,
                const float* __restrict__ W_lin, const float* __restrict__ b_lin,
                float* __restrict__ out)
{
    __shared__ __align__(64) _Float16 h_lds[KPAD];        // h (f16), zero-padded to K=128
    __shared__ __align__(64) float    gates_lds[NGATESPAD];
    __shared__ float                  hfin[HID];

    const int  tid   = threadIdx.x;
    const int  lane  = tid & 31;
    const int  wave  = tid >> 5;               // 0..12 : gate rows [32w, 32w+32)
    const bool upper = (lane & 16) != 0;       // lanes 16..31
    const int  mrow0 = wave * 32 + (lane & 15);
    const int  mrow1 = mrow0 + 16;

    // ---- One-time: load W_hh rows into WMMA A-layout f16 registers (2 tiles) ----
    // 16-bit A 16x32, this code's contiguous k order (B matches), lane m, VGPR v: k0 = 2v + (upper?16:0)
    v16h a0[4], a1[4];
    #pragma unroll
    for (int c = 0; c < 4; ++c) {
        #pragma unroll
        for (int v = 0; v < 8; ++v) {
            const int k0 = 32 * c + 2 * v + (upper ? 16 : 0);
            const bool k0ok = (k0 < HID), k1ok = (k0 + 1 < HID);
            float w00 = (k0ok && mrow0 < NGATES) ? W_hh[mrow0 * HID + k0]     : 0.0f;
            float w01 = (k1ok && mrow0 < NGATES) ? W_hh[mrow0 * HID + k0 + 1] : 0.0f;
            float w10 = (k0ok && mrow1 < NGATES) ? W_hh[mrow1 * HID + k0]     : 0.0f;
            float w11 = (k1ok && mrow1 < NGATES) ? W_hh[mrow1 * HID + k0 + 1] : 0.0f;
            a0[c][2 * v] = (_Float16)w00;  a0[c][2 * v + 1] = (_Float16)w01;
            a1[c][2 * v] = (_Float16)w10;  a1[c][2 * v + 1] = (_Float16)w11;
        }
    }

    // ---- One-time: per-hidden-unit constants for the cell update ----
    float c_state = 0.0f, h_state = 0.0f;
    float bi = 0.f, bf = 0.f, bg = 0.f, bo = 0.f;
    float wi = 0.f, wf = 0.f, wg = 0.f, wo = 0.f;
    if (tid < HID) {
        bi = b_ih[tid          ] + b_hh[tid          ];
        bf = b_ih[HID     + tid] + b_hh[HID     + tid];
        bg = b_ih[2 * HID + tid] + b_hh[2 * HID + tid];
        bo = b_ih[3 * HID + tid] + b_hh[3 * HID + tid];
        wi = W_ih[tid];           wf = W_ih[HID + tid];
        wg = W_ih[2 * HID + tid]; wo = W_ih[3 * HID + tid];
    }
    if (tid < KPAD) h_lds[tid] = (_Float16)0.0f;   // h0 = 0; padding stays 0 forever
    __syncthreads();

    float x_next = x_seq[0];
    const _Float16* bsrc = &h_lds[upper ? 16 : 0]; // per-lane B source (broadcast cols)

    for (int t = 0; t < T; ++t) {
        // ---- all four B chunks first: 8x ds_load_b128, ONE dscnt wait ----
        v16h b0 = *(const v16h*)(bsrc);
        v16h b1 = *(const v16h*)(bsrc + 32);
        v16h b2 = *(const v16h*)(bsrc + 64);
        v16h b3 = *(const v16h*)(bsrc + 96);
        __builtin_amdgcn_sched_barrier(0);   // pin all B loads ahead of the WMMA chain

        // ---- gates = W_hh @ h : two independent 4-deep WMMA chains ----
        v8f acc0 = {}, acc1 = {};
        acc0 = __builtin_amdgcn_wmma_f32_16x16x32_f16(false, a0[0], false, b0, (short)0, acc0, false, false);
        acc1 = __builtin_amdgcn_wmma_f32_16x16x32_f16(false, a1[0], false, b0, (short)0, acc1, false, false);
        acc0 = __builtin_amdgcn_wmma_f32_16x16x32_f16(false, a0[1], false, b1, (short)0, acc0, false, false);
        acc1 = __builtin_amdgcn_wmma_f32_16x16x32_f16(false, a1[1], false, b1, (short)0, acc1, false, false);
        acc0 = __builtin_amdgcn_wmma_f32_16x16x32_f16(false, a0[2], false, b2, (short)0, acc0, false, false);
        acc1 = __builtin_amdgcn_wmma_f32_16x16x32_f16(false, a1[2], false, b2, (short)0, acc1, false, false);
        acc0 = __builtin_amdgcn_wmma_f32_16x16x32_f16(false, a0[3], false, b3, (short)0, acc0, false, false);
        acc1 = __builtin_amdgcn_wmma_f32_16x16x32_f16(false, a1[3], false, b3, (short)0, acc1, false, false);

        // Columns are identical; lanes 0 and 16 publish both tiles' 16 rows each.
        if ((lane & 15) == 0) {
            float* g = &gates_lds[wave * 32 + (upper ? 8 : 0)];
            v4f t0lo = {acc0[0], acc0[1], acc0[2], acc0[3]};
            v4f t0hi = {acc0[4], acc0[5], acc0[6], acc0[7]};
            v4f t1lo = {acc1[0], acc1[1], acc1[2], acc1[3]};
            v4f t1hi = {acc1[4], acc1[5], acc1[6], acc1[7]};
            *(v4f*)(g)          = t0lo;
            *(v4f*)(g + 4)      = t0hi;
            *(v4f*)(g + 16)     = t1lo;
            *(v4f*)(g + 16 + 4) = t1hi;
        }
        __syncthreads();

        // ---- pointwise LSTM cell update (threads 0..99) ----
        if (tid < HID) {
            const float x = x_next;
            x_next = x_seq[(t + 1 < T) ? (t + 1) : t];   // scalar prefetch of next input
            const float gi = gates_lds[tid          ] + x * wi + bi;
            const float gf = gates_lds[HID     + tid] + x * wf + bf;
            const float gg = gates_lds[2 * HID + tid] + x * wg + bg;
            const float go = gates_lds[3 * HID + tid] + x * wo + bo;
            const float i_ = fsigmoid(gi);
            const float f_ = fsigmoid(gf);
            const float g_ = ftanh(gg);
            const float o_ = fsigmoid(go);
            c_state = f_ * c_state + i_ * g_;
            h_state = o_ * ftanh(c_state);
            h_lds[tid] = (_Float16)h_state;
        }
        __syncthreads();
    }

    // ---- final scalar: W_lin @ h_T + b_lin ----
    if (tid < HID) hfin[tid] = h_state;
    __syncthreads();
    if (tid == 0) {
        float s = b_lin[0];
        #pragma unroll 4
        for (int j = 0; j < HID; ++j) s += W_lin[j] * hfin[j];
        out[0] = s;
    }
}

extern "C" void kernel_launch(void* const* d_in, const int* in_sizes, int n_in,
                              void* d_out, int out_size, void* d_ws, size_t ws_size,
                              hipStream_t stream) {
    (void)n_in; (void)out_size; (void)d_ws; (void)ws_size;
    const float* x_seq = (const float*)d_in[0];   // [T]
    const float* W_ih  = (const float*)d_in[1];   // [400,1]
    const float* W_hh  = (const float*)d_in[2];   // [400,100]
    const float* b_ih  = (const float*)d_in[3];   // [400]
    const float* b_hh  = (const float*)d_in[4];   // [400]
    const float* W_lin = (const float*)d_in[5];   // [1,100]
    const float* b_lin = (const float*)d_in[6];   // [1]
    float*       outp  = (float*)d_out;           // [1]
    const int T = in_sizes[0];

    lstm_seq_kernel<<<dim3(1), dim3(NTHREADS), 0, stream>>>(
        x_seq, T, W_ih, W_hh, b_ih, b_hh, W_lin, b_lin, outp);
}
